// AttentionRPE_8383776162284
// MI455X (gfx1250) — matrix operand order repaired and checked by reading, not compiled, not run on hardware
//
#include <hip/hip_runtime.h>

// AttentionRPE fused kernel for MI455X (gfx1250, wave32, WMMA).
// D=256, H=8, DH=32, B=2, S=512, T=128, DR=32.
//
// Pipeline (all on `stream`):
//   1) convert_weights: w_tgt/rpe_weight f32->bf16; transpose w_src/out_proj_w (f32).
//   2) qproj: q = src @ w_src^T + b_src  (tiny, VALU, coalesced via transposed W).
//   3) attn_fused: one workgroup per (b,s):
//        - stage tgt tile (128x256) + rpe tile (128x32) f32->bf16 into LDS
//        - kv+rp projection via v_wmma_f32_16x16x32_bf16 (K=256 + K=32 RPE)
//          * wave w owns N-tiles {w, w+8, w+16, w+24}: B-frags held in VGPRs and
//            reused across all 8 M-tiles -> weights read from L2 ONCE per WG
//        - k-columns fused straight into attention logits (never stored)
//        - v-columns stored bf16 in LDS; softmax in LDS; outh; final out proj.

#define D_MODEL 256
#define NHEAD   8
#define DHEAD   32
#define BB      2
#define SS      512
#define TT      128
#define DRR     32
#define BSTOT   (BB * SS)          // 1024 workgroups

typedef __attribute__((ext_vector_type(16))) __bf16 v16bf;
typedef __attribute__((ext_vector_type(8)))  float  v8f;

union FragBF {
    struct { uint4 lo, hi; } u;    // 32 bytes
    v16bf v;
};

__device__ __forceinline__ unsigned short f2bf(float x) {
    unsigned int u = __float_as_uint(x);
    u += 0x7FFFu + ((u >> 16) & 1u);     // round-to-nearest-even
    return (unsigned short)(u >> 16);
}
__device__ __forceinline__ float bf2f(unsigned short b) {
    return __uint_as_float(((unsigned int)b) << 16);
}

// ---------------------------------------------------------------- kernel 0
__global__ void convert_weights_kernel(const float* __restrict__ in_proj_w,
                                       const float* __restrict__ rpe_w,
                                       const float* __restrict__ out_w,
                                       unsigned short* __restrict__ wtg_bf,
                                       unsigned short* __restrict__ rpw_bf,
                                       float* __restrict__ wout_t,
                                       float* __restrict__ wsrc_t) {
    int i = blockIdx.x * blockDim.x + threadIdx.x;
    int stride = gridDim.x * blockDim.x;
    // w_tgt = in_proj_weight[256:768]  (512 x 256) -> bf16
    for (int idx = i; idx < 512 * 256; idx += stride)
        wtg_bf[idx] = f2bf(in_proj_w[256 * 256 + idx]);
    // rpe_weight (512 x 32) -> bf16
    for (int idx = i; idx < 512 * 32; idx += stride)
        rpw_bf[idx] = f2bf(rpe_w[idx]);
    // transposed f32 copies for coalesced GEMV loops:
    //   wout_t[e*256 + dout] = out_w[dout*256 + e]
    //   wsrc_t[d*256 + dout] = in_proj_w[dout*256 + d]   (w_src rows 0..255)
    for (int idx = i; idx < 256 * 256; idx += stride) {
        wout_t[idx] = out_w[(idx & 255) * 256 + (idx >> 8)];
        wsrc_t[idx] = in_proj_w[(idx & 255) * 256 + (idx >> 8)];
    }
}

// ---------------------------------------------------------------- kernel 1
// q = src @ w_src^T + b_src ; 1024 blocks x 256 threads, one row per block.
__global__ void qproj_kernel(const float* __restrict__ src,
                             const float* __restrict__ wsrc_t,  // [d][dout]
                             const float* __restrict__ in_proj_b,
                             float* __restrict__ q_ws) {
    __shared__ float s[256];
    const int bs = blockIdx.x, tid = threadIdx.x;
    s[tid] = src[(size_t)bs * 256 + tid];
    __syncthreads();
    float acc = in_proj_b[tid];
    #pragma unroll 8
    for (int d = 0; d < 256; ++d) acc += s[d] * wsrc_t[d * 256 + tid];
    q_ws[(size_t)bs * 256 + tid] = acc;
}

// ---------------------------------------------------------------- kernel 2
#define TGT_LSTR 264   // bf16 elems per row (256 + pad, mult of 8)
#define RPE_LSTR 40
#define VS_LSTR  264
#define SMEM_BYTES ((128 * TGT_LSTR + 128 * RPE_LSTR + 128 * VS_LSTR) * 2 \
                    + (8 * 128 + 256 + 256) * 4 + 16)

__global__ __launch_bounds__(256)
void attn_fused_kernel(const float* __restrict__ tgt,
                       const float* __restrict__ rpe,
                       const unsigned char* __restrict__ maskp,   // jax bool: 1 byte
                       const float* __restrict__ q_ws,
                       const unsigned short* __restrict__ wtg,    // bf16 w_tgt 512x256
                       const unsigned short* __restrict__ rpw,    // bf16 rpe_w 512x32
                       const float* __restrict__ in_proj_b,
                       const float* __restrict__ rpe_b,
                       const float* __restrict__ wout_t,          // f32 [e][dout]
                       const float* __restrict__ out_b,
                       float* __restrict__ out) {
    extern __shared__ char smem_raw[];
    unsigned short* tgt_s  = (unsigned short*)smem_raw;            // 128 x TGT_LSTR
    unsigned short* rpe_s  = tgt_s + 128 * TGT_LSTR;               // 128 x RPE_LSTR
    unsigned short* vsum_s = rpe_s + 128 * RPE_LSTR;               // 128 x VS_LSTR
    float* logits = (float*)(vsum_s + 128 * VS_LSTR);              // 8 x 128
    float* q_s    = logits + 8 * 128;                              // 256
    float* outh_s = q_s + 256;                                     // 256
    int*   nv_s   = (int*)(outh_s + 256);

    const int tid = threadIdx.x;
    const int bs  = blockIdx.x;          // (b*S + s)

    // ---- Phase 0: stage tgt/rpe tiles (f32 -> bf16) + q, zero logits ----
    {
        const float4* t4 = (const float4*)(tgt + (size_t)bs * TT * D_MODEL);
        #pragma unroll
        for (int i = 0; i < 32; ++i) {
            int idx = tid + i * 256;          // float4 index, row-major contiguous
            float4 v = t4[idx];
            int row = idx >> 6, c = (idx & 63) << 2;
            unsigned short* p = tgt_s + row * TGT_LSTR + c;
            p[0] = f2bf(v.x); p[1] = f2bf(v.y); p[2] = f2bf(v.z); p[3] = f2bf(v.w);
        }
        const float4* r4 = (const float4*)(rpe + (size_t)bs * TT * DRR);
        #pragma unroll
        for (int i = 0; i < 4; ++i) {
            int idx = tid + i * 256;
            float4 v = r4[idx];
            int row = idx >> 3, c = (idx & 7) << 2;
            unsigned short* p = rpe_s + row * RPE_LSTR + c;
            p[0] = f2bf(v.x); p[1] = f2bf(v.y); p[2] = f2bf(v.z); p[3] = f2bf(v.w);
        }
        q_s[tid] = q_ws[(size_t)bs * 256 + tid];
        #pragma unroll
        for (int i = 0; i < 4; ++i) logits[tid + i * 256] = 0.0f;
    }
    __syncthreads();

    const int wave  = tid >> 5;
    const int lane  = tid & 31;
    const int n     = lane & 15;      // N-column / M-row within half
    const int khalf = lane >> 4;

    // ---- Phase 1: (kv + rp) projection via WMMA bf16 ----
    // Wave w owns N-tiles nt in {w, w+8, w+16, w+24}. B-frags (9 x 32B = 72 VGPRs)
    // are loaded from global ONCE per nt and reused across all 8 M-tiles; A-frags
    // are re-read from the LDS tgt/rpe tiles (two ds_load_b128 each).
    for (int i = 0; i < 4; ++i) {
        const int nt = wave + i * 8;
        const int e0 = nt * 16;
        const int e_col = e0 + n;                 // output column (0..511)

        // B-frag (32x16 bf16): lane(n, khalf), elems e: K = kb*32 + khalf*16 + e
        FragBF b[8], br;
        const unsigned short* wrow = wtg + (size_t)e_col * 256 + khalf * 16;
        #pragma unroll
        for (int kb = 0; kb < 8; ++kb) {
            b[kb].u.lo = *(const uint4*)(wrow + kb * 32);
            b[kb].u.hi = *(const uint4*)(wrow + kb * 32 + 8);
        }
        const unsigned short* rrow = rpw + (size_t)e_col * 32 + khalf * 16;
        br.u.lo = *(const uint4*)(rrow);
        br.u.hi = *(const uint4*)(rrow + 8);

        const float bias = in_proj_b[256 + e_col] + rpe_b[e_col];
        const bool is_k = (e0 < 256);
        const int h  = nt >> 1;
        const int vb = e0 - 256;
        const float qv = is_k ? q_s[e_col] : 0.0f;

        for (int mt = 0; mt < 8; ++mt) {
            // A-frag (16x32 bf16): lane(m=n, khalf),
            // elems e: K = kb*32 + khalf*16 + e
            const unsigned short* arow = tgt_s + (mt * 16 + n) * TGT_LSTR + khalf * 16;
            v8f acc = {};
            #pragma unroll
            for (int kb = 0; kb < 8; ++kb) {
                FragBF a;
                a.u.lo = *(const uint4*)(arow + kb * 32);
                a.u.hi = *(const uint4*)(arow + kb * 32 + 8);
                acc = __builtin_amdgcn_wmma_f32_16x16x32_bf16(
                    false, a.v, false, b[kb].v, (short)0, acc, false, false);
            }
            {   // RPE projection contribution (K = 32)
                FragBF ar;
                const unsigned short* rp = rpe_s + (mt * 16 + n) * RPE_LSTR + khalf * 16;
                ar.u.lo = *(const uint4*)(rp);
                ar.u.hi = *(const uint4*)(rp + 8);
                acc = __builtin_amdgcn_wmma_f32_16x16x32_bf16(
                    false, ar.v, false, br.v, (short)0, acc, false, false);
            }
            #pragma unroll
            for (int j = 0; j < 8; ++j) acc[j] += bias;

            if (is_k) {
                // k columns: fuse attention logits q . (k + rpe_k), never store k.
                // Each logits[h][t] gets exactly 2 atomic adds (nt=2h, nt=2h+1)
                // onto 0.0 -> commutative -> bitwise deterministic.
                #pragma unroll
                for (int r = 0; r < 8; ++r) {
                    float p = qv * acc[r];
                    p += __shfl_xor(p, 1, 32);
                    p += __shfl_xor(p, 2, 32);
                    p += __shfl_xor(p, 4, 32);
                    p += __shfl_xor(p, 8, 32);    // sum over 16 N-lanes per half
                    if (n == 0)
                        atomicAdd(&logits[h * 128 + mt * 16 + khalf * 8 + r], p);
                }
            } else {
                // v columns: keep (v + rpe_v) as bf16 in LDS for post-softmax use.
                #pragma unroll
                for (int r = 0; r < 8; ++r)
                    vsum_s[(mt * 16 + khalf * 8 + r) * VS_LSTR + vb + n] = f2bf(acc[r]);
            }
        }
    }
    __syncthreads();

    // ---- Phase 2: masked softmax, wave w handles head w (128 t over 32 lanes) ----
    {
        const int h = wave;
        const unsigned char* mrow = maskp + (size_t)bs * TT;
        bool  m[4];
        float lg[4];
        bool allm = true;
        #pragma unroll
        for (int j = 0; j < 4; ++j) {
            int t = lane + j * 32;
            m[j] = mrow[t] != 0;
            allm = allm && m[j];
        }
        unsigned av = allm ? 1u : 0u;
        av &= __shfl_xor(av, 1, 32);
        av &= __shfl_xor(av, 2, 32);
        av &= __shfl_xor(av, 4, 32);
        av &= __shfl_xor(av, 8, 32);
        av &= __shfl_xor(av, 16, 32);
        const bool no_valid = (av != 0);          // all positions padded
        const float scale = 0.17677669529663687f; // 1/sqrt(DH)
        float mx = -3.0e38f;
        #pragma unroll
        for (int j = 0; j < 4; ++j) {
            int t = lane + j * 32;
            bool eff = m[j] && !no_valid;         // mask & ~no_valid
            lg[j] = eff ? -3.0e38f : logits[h * 128 + t] * scale;
            mx = fmaxf(mx, lg[j]);
        }
        mx = fmaxf(mx, __shfl_xor(mx, 1, 32));
        mx = fmaxf(mx, __shfl_xor(mx, 2, 32));
        mx = fmaxf(mx, __shfl_xor(mx, 4, 32));
        mx = fmaxf(mx, __shfl_xor(mx, 8, 32));
        mx = fmaxf(mx, __shfl_xor(mx, 16, 32));
        float se = 0.0f;
        #pragma unroll
        for (int j = 0; j < 4; ++j) {
            lg[j] = __expf(lg[j] - mx);
            se += lg[j];
        }
        se += __shfl_xor(se, 1, 32);
        se += __shfl_xor(se, 2, 32);
        se += __shfl_xor(se, 4, 32);
        se += __shfl_xor(se, 8, 32);
        se += __shfl_xor(se, 16, 32);
        const float inv = 1.0f / se;
        #pragma unroll
        for (int j = 0; j < 4; ++j)
            logits[h * 128 + lane + j * 32] = lg[j] * inv;   // attn weights in place
        if (tid == 0) *nv_s = no_valid ? 1 : 0;
    }
    __syncthreads();

    // ---- Phase 3: outh[h][dh] = sum_t attn[h][t] * (v+rpe_v)[t][h*32+dh] ----
    {
        const float* arow = logits + (tid >> 5) * 128;   // col = tid = h*32+dh
        float acc = 0.0f;
        #pragma unroll 4
        for (int t = 0; t < 128; ++t)
            acc += arow[t] * bf2f(vsum_s[t * VS_LSTR + tid]);
        outh_s[tid] = acc;
    }
    __syncthreads();

    // ---- Phase 4: out = outh @ out_proj_w^T + b ; zero if no_valid ----
    {
        float acc = out_b[tid];
        #pragma unroll 8
        for (int e = 0; e < 256; ++e) acc += outh_s[e] * wout_t[e * 256 + tid];
        out[(size_t)bs * 256 + tid] = (*nv_s) ? 0.0f : acc;
    }
}

// ---------------------------------------------------------------- launch
extern "C" void kernel_launch(void* const* d_in, const int* in_sizes, int n_in,
                              void* d_out, int out_size, void* d_ws, size_t ws_size,
                              hipStream_t stream) {
    (void)in_sizes; (void)n_in; (void)out_size; (void)ws_size;
    const float* src  = (const float*)d_in[0];
    const float* tgt  = (const float*)d_in[1];
    const float* rpe  = (const float*)d_in[2];
    const unsigned char* mask = (const unsigned char*)d_in[3];  // jax bool = 1B
    const float* ipw  = (const float*)d_in[4];   // (768, 256)
    const float* ipb  = (const float*)d_in[5];   // (768,)
    const float* opw  = (const float*)d_in[6];   // (256, 256)
    const float* opb  = (const float*)d_in[7];   // (256,)
    const float* rpw  = (const float*)d_in[8];   // (512, 32)
    const float* rpb  = (const float*)d_in[9];   // (512,)

    // Workspace: [0,1MB) q f32 ; [1MB) w_tgt bf16 (256KB) ; [1.25MB) rpe_w bf16
    //            (32KB) ; [1.5MB) out_w^T f32 (256KB) ; [1.75MB) w_src^T f32 (256KB).
    float* q_ws = (float*)d_ws;
    unsigned short* wtg_bf = (unsigned short*)((char*)d_ws + (1u << 20));
    unsigned short* rpw_bf = (unsigned short*)((char*)d_ws + (1u << 20) + (256u << 10));
    float* wout_t = (float*)((char*)d_ws + (3u << 19));          // 1.5 MB
    float* wsrc_t = (float*)((char*)d_ws + (7u << 18));          // 1.75 MB

    convert_weights_kernel<<<256, 256, 0, stream>>>(ipw, rpw, opw,
                                                    wtg_bf, rpw_bf, wout_t, wsrc_t);
    qproj_kernel<<<BSTOT, 256, 0, stream>>>(src, wsrc_t, ipb, q_ws);
    attn_fused_kernel<<<BSTOT, 256, SMEM_BYTES, stream>>>(
        tgt, rpe, mask, q_ws, wtg_bf, rpw_bf, ipb, rpb, wout_t, opb, (float*)d_out);
}
